// NLManager_8263517077807
// MI455X (gfx1250) — matrix operand order repaired and checked by reading, not compiled, not run on hardware
//
#include <hip/hip_runtime.h>

// ---------------------------------------------------------------------------
// CDNA5 / gfx1250 fused nested-logit manager.
//   Phase 0: swizzle w1,w2 fp32 -> bf16 WMMA B-fragments in workspace
//   Phase 1: fused MLP (x@w1 -> relu -> @w2 -> relu -> @w3) via
//            v_wmma_f32_16x16x32_bf16, 32-row tiles (2 A-tiles per wave,
//            each B-fragment reused for 2 WMMAs -> 2x arithmetic intensity
//            vs L2), utils -> d_out
//   Phase 2: per-batch nested softmax (M=4 nests) -> p_task, p_nest
// ---------------------------------------------------------------------------

typedef __attribute__((ext_vector_type(16))) __bf16        v16bf;
typedef __attribute__((ext_vector_type(8)))  float         v8f;
typedef __attribute__((ext_vector_type(4)))  unsigned int  v4u;
typedef __attribute__((ext_vector_type(8)))  unsigned int  v8u;

static constexpr int B_  = 128;
static constexpr int T_  = 512;
static constexpr int S_  = 512;
static constexpr int E_  = 256;
static constexpr int H_  = 1024;
static constexpr int M_  = 4;
static constexpr int K1_ = S_ + E_;   // 768
static constexpr int RT_ = 32;        // rows per workgroup tile

#define NEGV (-1.0e9f)
#define EPSV (1e-12f)

__device__ __forceinline__ unsigned short f2bf(float f) {
  unsigned int u = __builtin_bit_cast(unsigned int, f);
  u += 0x7FFFu + ((u >> 16) & 1u);          // round-to-nearest-even
  return (unsigned short)(u >> 16);
}
__device__ __forceinline__ float bf2f(unsigned short s) {
  unsigned int u = ((unsigned int)s) << 16;
  return __builtin_bit_cast(float, u);
}

// A-fragment (16x32 bf16, M x K) from an LDS row-major bf16 tile.
// Per ISA 7.12.2: lane<16 holds K = {k0..k0+7, k0+16..k0+23} with k0=0,
// lane>=16 same with k0=8; row M = lane%16.  Two 16-byte LDS loads.
__device__ __forceinline__ v16bf ld_a_frag(const unsigned short* lds, int row,
                                           int rowStride, int kBase, int lane) {
  int k0 = kBase + ((lane & 16) ? 8 : 0);
  const unsigned short* p = lds + row * rowStride + k0;
  v4u lo = *reinterpret_cast<const v4u*>(p);
  v4u hi = *reinterpret_cast<const v4u*>(p + 16);
  v8u u;
  u[0] = lo[0]; u[1] = lo[1]; u[2] = lo[2]; u[3] = lo[3];
  u[4] = hi[0]; u[5] = hi[1]; u[6] = hi[2]; u[7] = hi[3];
  return __builtin_bit_cast(v16bf, u);
}

// B-fragment (32x16 bf16, K x N) pre-swizzled in global memory:
// each lane reads 32 contiguous bytes, coalesced across the wave.
__device__ __forceinline__ v16bf ld_b_frag(const unsigned short* wbf,
                                           int fragIdx, int lane) {
  const v8u* p = reinterpret_cast<const v8u*>(wbf);
  v8u u = p[fragIdx * 32 + lane];
  return __builtin_bit_cast(v16bf, u);
}

// ---------------------------------------------------------------------------
// Phase 0: fp32 weights -> bf16 B-fragments.
// Fragment layout: frag = kt*(N/16)+nt ; lane l : N = nt*16 + l%16,
// K = kt*32 + (l<16 ? 0 : 8) + e + (e<8 ? 0 : 8), e = 0..15 contiguous in memory.
// ---------------------------------------------------------------------------
__global__ void swizzle_weights(const float* __restrict__ w,
                                unsigned short* __restrict__ out,
                                int K, int N) {
  int idx = blockIdx.x * blockDim.x + threadIdx.x;
  int total = (K / 32) * (N / 16) * 32;
  if (idx >= total) return;
  int lane = idx & 31;
  int frag = idx >> 5;
  int nNT = N / 16;
  int kt = frag / nNT, nt = frag % nNT;
  int n  = nt * 16 + (lane & 15);
  int kb = kt * 32 + ((lane & 16) ? 8 : 0);
  unsigned short* o = out + idx * 16;
#pragma unroll
  for (int e = 0; e < 16; ++e) o[e] = f2bf(w[(kb + e + (e < 8 ? 0 : 8)) * N + n]);
}

// ---------------------------------------------------------------------------
// Phase 1: fused MLP, one 32-row tile per workgroup (8 waves, wave32).
// Each wave owns 128 of the 1024 hidden columns; each B-fragment is
// loaded once and used for both 16-row A-tiles.
// ---------------------------------------------------------------------------
__global__ __launch_bounds__(256) void mlp_utils_kernel(
    const float* __restrict__ state, const float* __restrict__ tf,
    const unsigned char* __restrict__ mask,
    const float* __restrict__ emb_w, const float* __restrict__ emb_b,
    const unsigned short* __restrict__ w1bf, const float* __restrict__ b1,
    const unsigned short* __restrict__ w2bf, const float* __restrict__ b2,
    const float* __restrict__ w3, const float* __restrict__ b3,
    float* __restrict__ utils_out) {

  __shared__ __align__(16) unsigned short buf0[RT_ * 1024]; // x (stride 768), reused for h2 (stride 1024)
  __shared__ __align__(16) unsigned short buf1[RT_ * 1024]; // h1 (stride 1024)
  __shared__ float red[RT_][8];

  const int tid  = threadIdx.x;
  const int lane = tid & 31;
  const int wave = tid >> 5;
  const int b    = blockIdx.x >> 4;        // T_/RT_ = 16 tiles per batch row
  const int t0   = (blockIdx.x & 15) * RT_;

  // ---- build x tile [32 x 768] bf16 ----
  for (int i = tid; i < RT_ * S_; i += 256) {
    int r = i >> 9, c = i & (S_ - 1);
    buf0[r * K1_ + c] = f2bf(state[b * S_ + c]);
  }
  for (int i = tid; i < RT_ * E_; i += 256) {
    int r = i >> 8, e = i & (E_ - 1);
    int t = t0 + r;
    float acc = emb_b[e];
#pragma unroll
    for (int j = 0; j < 5; ++j)
      acc += tf[(b * T_ + t) * 5 + j] * emb_w[j * E_ + e];
    buf0[r * K1_ + S_ + e] = f2bf(acc);
  }
  __syncthreads();

  // ---- layer 1: h1 = relu(x @ w1 + b1) ----
  {
    v8f acc0[8] = {};   // rows 0..15
    v8f acc1[8] = {};   // rows 16..31
    for (int kt = 0; kt < K1_ / 32; ++kt) {
      v16bf a0 = ld_a_frag(buf0,             lane & 15, K1_, kt * 32, lane);
      v16bf a1 = ld_a_frag(buf0 + 16 * K1_,  lane & 15, K1_, kt * 32, lane);
#pragma unroll
      for (int j = 0; j < 8; ++j) {
        int nt = wave * 8 + j;
        v16bf bf = ld_b_frag(w1bf, kt * (H_ / 16) + nt, lane);
        acc0[j] = __builtin_amdgcn_wmma_f32_16x16x32_bf16(
            false, a0, false, bf, (short)0, acc0[j], false, false);
        acc1[j] = __builtin_amdgcn_wmma_f32_16x16x32_bf16(
            false, a1, false, bf, (short)0, acc1[j], false, false);
      }
    }
    int rbase = (lane & 16) ? 8 : 0;
#pragma unroll
    for (int j = 0; j < 8; ++j) {
      int col = (wave * 8 + j) * 16 + (lane & 15);
      float bias = b1[col];
#pragma unroll
      for (int v = 0; v < 8; ++v) {
        float h0 = acc0[j][v] + bias;
        float h1 = acc1[j][v] + bias;
        buf1[(rbase + v) * H_ + col]        = f2bf(h0 > 0.0f ? h0 : 0.0f);
        buf1[(16 + rbase + v) * H_ + col]   = f2bf(h1 > 0.0f ? h1 : 0.0f);
      }
    }
  }
  __syncthreads();

  // ---- layer 2: h2 = relu(h1 @ w2 + b2), into buf0 (x is dead) ----
  {
    v8f acc0[8] = {};
    v8f acc1[8] = {};
    for (int kt = 0; kt < H_ / 32; ++kt) {
      v16bf a0 = ld_a_frag(buf1,            lane & 15, H_, kt * 32, lane);
      v16bf a1 = ld_a_frag(buf1 + 16 * H_,  lane & 15, H_, kt * 32, lane);
#pragma unroll
      for (int j = 0; j < 8; ++j) {
        int nt = wave * 8 + j;
        v16bf bf = ld_b_frag(w2bf, kt * (H_ / 16) + nt, lane);
        acc0[j] = __builtin_amdgcn_wmma_f32_16x16x32_bf16(
            false, a0, false, bf, (short)0, acc0[j], false, false);
        acc1[j] = __builtin_amdgcn_wmma_f32_16x16x32_bf16(
            false, a1, false, bf, (short)0, acc1[j], false, false);
      }
    }
    int rbase = (lane & 16) ? 8 : 0;
#pragma unroll
    for (int j = 0; j < 8; ++j) {
      int col = (wave * 8 + j) * 16 + (lane & 15);
      float bias = b2[col];
#pragma unroll
      for (int v = 0; v < 8; ++v) {
        float h0 = acc0[j][v] + bias;
        float h1 = acc1[j][v] + bias;
        buf0[(rbase + v) * H_ + col]      = f2bf(h0 > 0.0f ? h0 : 0.0f);
        buf0[(16 + rbase + v) * H_ + col] = f2bf(h1 > 0.0f ? h1 : 0.0f);
      }
    }
  }
  __syncthreads();

  // ---- utils = h2 @ w3 + b3, masked ----
  {
    int r = tid >> 3, p = tid & 7;
    float s = 0.0f;
    for (int k = p * 128; k < p * 128 + 128; ++k)
      s += bf2f(buf0[r * H_ + k]) * w3[k];
    red[r][p] = s;
  }
  __syncthreads();
  if (tid < RT_) {
    float s = 0.0f;
#pragma unroll
    for (int p = 0; p < 8; ++p) s += red[tid][p];
    s += b3[0];
    int g = b * T_ + t0 + tid;
    utils_out[g] = mask[g] ? s : NEGV;
  }
}

// ---------------------------------------------------------------------------
// Phase 2: nested-logit softmax, one workgroup per batch row.
// ---------------------------------------------------------------------------
__global__ __launch_bounds__(256) void nested_softmax_kernel(
    const float* __restrict__ utils, const int* __restrict__ nest_ids,
    const unsigned char* __restrict__ mask, const float* __restrict__ etas,
    float* __restrict__ p_task, float* __restrict__ p_nest) {
  __shared__ float rbuf[256 * 4];
  __shared__ float s_mval[4], s_sum[4], s_pn[4], s_eta[4];
  __shared__ int   s_ne[4];
  __shared__ float s_uni;
  __shared__ int   s_fb;

  const int tid = threadIdx.x;
  const int b   = blockIdx.x;
  if (tid < 4) s_eta[tid] = etas[tid];
  __syncthreads();

  // per-nest max
  float lmax[4] = {-INFINITY, -INFINITY, -INFINITY, -INFINITY};
  for (int t = tid; t < T_; t += 256) {
    int g = b * T_ + t;
    int nid = nest_ids[g];
    if (mask[g] && nid >= 0 && nid < M_)
      lmax[nid] = fmaxf(lmax[nid], utils[g]);
  }
#pragma unroll
  for (int m = 0; m < 4; ++m) rbuf[tid * 4 + m] = lmax[m];
  __syncthreads();
  for (int s = 128; s > 0; s >>= 1) {
    if (tid < s)
#pragma unroll
      for (int m = 0; m < 4; ++m)
        rbuf[tid * 4 + m] = fmaxf(rbuf[tid * 4 + m], rbuf[(tid + s) * 4 + m]);
    __syncthreads();
  }
  if (tid < 4) {
    float mv = rbuf[tid];
    int ne = (mv != -INFINITY) ? 1 : 0;
    s_ne[tid] = ne;
    s_mval[tid] = ne ? mv : 0.0f;
  }
  __syncthreads();

  // per-nest sum of exp((u - mval)/eta)
  float lsum[4] = {0.0f, 0.0f, 0.0f, 0.0f};
  for (int t = tid; t < T_; t += 256) {
    int g = b * T_ + t;
    int nid = nest_ids[g];
    if (mask[g] && nid >= 0 && nid < M_)
      lsum[nid] += __expf((utils[g] - s_mval[nid]) / s_eta[nid]);
  }
#pragma unroll
  for (int m = 0; m < 4; ++m) rbuf[tid * 4 + m] = lsum[m];
  __syncthreads();
  for (int s = 128; s > 0; s >>= 1) {
    if (tid < s)
#pragma unroll
      for (int m = 0; m < 4; ++m)
        rbuf[tid * 4 + m] += rbuf[(tid + s) * 4 + m];
    __syncthreads();
  }
  if (tid < 4) {
    float sm = rbuf[tid];
    s_sum[tid] = sm < EPSV ? EPSV : sm;
  }
  __syncthreads();

  // inclusive values + nest softmax
  if (tid == 0) {
    float U[4], um = -INFINITY;
#pragma unroll
    for (int m = 0; m < 4; ++m) {
      U[m] = s_ne[m] ? s_mval[m] + s_eta[m] * __logf(s_sum[m]) : NEGV;
      um = fmaxf(um, U[m]);
    }
    float e[4], tot = 0.0f;
#pragma unroll
    for (int m = 0; m < 4; ++m) { e[m] = __expf(U[m] - um); tot += e[m]; }
#pragma unroll
    for (int m = 0; m < 4; ++m) {
      s_pn[m] = e[m] / tot;
      p_nest[b * M_ + m] = s_pn[m];
    }
  }
  __syncthreads();

  // per-task probabilities + fallback detection
  float lpt = 0.0f;
  int lcnt = 0;
  for (int t = tid; t < T_; t += 256) {
    int g = b * T_ + t;
    int nid = nest_ids[g];
    bool mk = mask[g] != 0;
    bool valid = mk && nid >= 0 && nid < M_;
    float pt = 0.0f;
    if (valid) {
      float e = __expf((utils[g] - s_mval[nid]) / s_eta[nid]);
      pt = s_pn[nid] * e / s_sum[nid];
    }
    p_task[g] = pt;
    lpt += pt;
    lcnt += mk ? 1 : 0;
  }
  rbuf[tid] = lpt;
  rbuf[256 + tid] = (float)lcnt;
  __syncthreads();
  for (int s = 128; s > 0; s >>= 1) {
    if (tid < s) {
      rbuf[tid] += rbuf[tid + s];
      rbuf[256 + tid] += rbuf[256 + tid + s];
    }
    __syncthreads();
  }
  if (tid == 0) {
    float ptsum = rbuf[0];
    int cnt = (int)rbuf[256];
    s_fb = (cnt > 0) && (ptsum <= EPSV);
    s_uni = 1.0f / (float)(cnt > 0 ? cnt : 1);
  }
  __syncthreads();
  if (s_fb) {
    for (int t = tid; t < T_; t += 256) {
      int g = b * T_ + t;
      p_task[g] = mask[g] ? s_uni : 0.0f;
    }
  }
}

// ---------------------------------------------------------------------------
extern "C" void kernel_launch(void* const* d_in, const int* in_sizes, int n_in,
                              void* d_out, int out_size, void* d_ws, size_t ws_size,
                              hipStream_t stream) {
  const float* state = (const float*)d_in[0];
  const float* tf    = (const float*)d_in[1];
  const int*   nid   = (const int*)d_in[2];
  const unsigned char* mask = (const unsigned char*)d_in[3];  // jnp.bool_ = 1 byte
  const float* emb_w = (const float*)d_in[4];
  const float* emb_b = (const float*)d_in[5];
  const float* w1    = (const float*)d_in[6];
  const float* b1    = (const float*)d_in[7];
  const float* w2    = (const float*)d_in[8];
  const float* b2    = (const float*)d_in[9];
  const float* w3    = (const float*)d_in[10];
  const float* b3    = (const float*)d_in[11];
  const float* etas  = (const float*)d_in[12];

  unsigned short* w1bf = (unsigned short*)d_ws;                 // 768*1024 bf16
  unsigned short* w2bf = w1bf + (size_t)K1_ * H_;               // 1024*1024 bf16

  float* utils  = (float*)d_out;
  float* p_task = utils + B_ * T_;
  float* p_nest = p_task + B_ * T_;

  {
    int total = (K1_ / 32) * (H_ / 16) * 32;
    swizzle_weights<<<(total + 255) / 256, 256, 0, stream>>>(w1, w1bf, K1_, H_);
  }
  {
    int total = (H_ / 32) * (H_ / 16) * 32;
    swizzle_weights<<<(total + 255) / 256, 256, 0, stream>>>(w2, w2bf, H_, H_);
  }
  mlp_utils_kernel<<<B_ * (T_ / RT_), 256, 0, stream>>>(
      state, tf, mask, emb_w, emb_b, w1bf, b1, w2bf, b2, w3, b3, utils);
  nested_softmax_kernel<<<B_, 256, 0, stream>>>(utils, nid, mask, etas,
                                                p_task, p_nest);
}
